// CausalAttention_18451179504230
// MI455X (gfx1250) — compile-verified, matrix operand order checked by reading, not run on hardware
//
#include <hip/hip_runtime.h>
#include <hip/hip_bf16.h>

// Problem constants (match reference)
constexpr int BB  = 4;     // batch
constexpr int SEQ = 2048;  // T
constexpr int DIM = 2048;  // D
constexpr int NH  = 16;    // heads
constexpr int DHD = 128;   // head dim

typedef __bf16 v16bf __attribute__((ext_vector_type(16)));
typedef float  v8f   __attribute__((ext_vector_type(8)));
typedef int    v4i   __attribute__((vector_size(16)));  // matches builtin sig

// ---------------------------------------------------------------------------
// Async global->LDS (GLOBAL_LOAD_ASYNC_TO_LDS_B128, ASYNCcnt-tracked).
// Builtin signature (from compiler diagnostic): param 0 is AS(1) v4i*.
// Host pass / older toolchains fall back to the synchronous VGPR-staged path.
// ---------------------------------------------------------------------------
#if defined(__has_builtin)
#  if __has_builtin(__builtin_amdgcn_global_load_async_to_lds_b128)
#    define HAVE_ASYNC_LDS 1
#  endif
#endif
#ifndef HAVE_ASYNC_LDS
#  define HAVE_ASYNC_LDS 0
#endif

#if HAVE_ASYNC_LDS
__device__ inline void async_copy16(const void* g, void* l) {
  __builtin_amdgcn_global_load_async_to_lds_b128(
      (__attribute__((address_space(1))) v4i*)g,
      (__attribute__((address_space(3))) v4i*)l, 0, 0);
}
#endif

template <int N>
__device__ inline void wait_async() {
#if defined(__has_builtin)
#  if __has_builtin(__builtin_amdgcn_s_wait_asynccnt)
  __builtin_amdgcn_s_wait_asynccnt(N);
#  else
  asm volatile("s_wait_asynccnt %0" ::"n"(N) : "memory");
#  endif
#else
  (void)N;
#endif
}

// ---------------------------------------------------------------------------
// Load a 16x32 bf16 operand fragment (A layout per ISA 7.12.2).
// Per lane: row = lane%16, half = lane/16.
//   VGPRs 0..3  <- K = koff + half*8 + [0..7]      (one b128)
//   VGPRs 4..7  <- K = koff + 16 + half*8 + [0..7] (one b128)
// B operand (KxN) is loaded with the same routine from a B^T (NxK) tile.
// ---------------------------------------------------------------------------
__device__ inline v16bf load_frag(const __bf16* base, int stride, int koff) {
  const int lane = threadIdx.x & 31;
  const __bf16* p = base + (size_t)(lane & 15) * stride + koff + ((lane >> 4) << 3);
  union { uint4 u[2]; v16bf v; } r;
  r.u[0] = *(const uint4*)(p);
  r.u[1] = *(const uint4*)(p + 16);
  return r.v;
}

__device__ inline v8f wmma_bf16(v16bf a, v16bf b, v8f c) {
  return __builtin_amdgcn_wmma_f32_16x16x32_bf16(false, a, false, b,
                                                 (short)0, c, false, false);
}

// ---------------------------------------------------------------------------
// fp32 -> bf16 elementwise (8 elements / thread, 16B stores)
// ---------------------------------------------------------------------------
__global__ void __launch_bounds__(256)
cvt_kernel(const float* __restrict__ src, __bf16* __restrict__ dst) {
  size_t i = ((size_t)blockIdx.x * blockDim.x + threadIdx.x) * 8;
  float4 a = *(const float4*)(src + i);
  float4 b = *(const float4*)(src + i + 4);
  __bf16 o[8] = {(__bf16)a.x, (__bf16)a.y, (__bf16)a.z, (__bf16)a.w,
                 (__bf16)b.x, (__bf16)b.y, (__bf16)b.z, (__bf16)b.w};
  *(uint4*)(dst + i) = *(const uint4*)o;
}

// ---------------------------------------------------------------------------
// fp32 [K,N] -> bf16 [N,K] (weight transpose so GEMM B-operand is K-contiguous)
// ---------------------------------------------------------------------------
__global__ void __launch_bounds__(256)
transpose_cvt_kernel(const float* __restrict__ W, __bf16* __restrict__ Wt) {
  __shared__ float tile[32][33];
  const int bx = blockIdx.x * 32, by = blockIdx.y * 32;
  const int x = threadIdx.x, y = threadIdx.y;
  #pragma unroll
  for (int i = 0; i < 32; i += 8)
    tile[y + i][x] = W[(size_t)(by + y + i) * DIM + bx + x];
  __syncthreads();
  #pragma unroll
  for (int i = 0; i < 32; i += 8)
    Wt[(size_t)(bx + y + i) * DIM + by + x] = (__bf16)tile[x][y + i];
}

// ---------------------------------------------------------------------------
// GEMM  C[M,N] = A[M,K] * Bt[N,K]^T + bias
//   MODE 0: store bf16, head-split   [B,H,T,DH]   (Q, K projections)
//   MODE 1: store bf16, head-split-T [B,H,DH,T]   (V projection)
//   MODE 2: store fp32 flat [M,N]                 (output projection)
// Block: 256 threads = 8 waves; 128x128 C tile; BK=32; wave tile 32x64.
// Async path: double-buffered LDS, next tile in flight while WMMAs run.
// ---------------------------------------------------------------------------
constexpr int GBK = 32;
constexpr int GLSA = GBK + 8;  // 40 elem = 80B rows (16B aligned)

#if HAVE_ASYNC_LDS
__device__ inline void gemm_issue_tile(const __bf16* __restrict__ A,
                                       const __bf16* __restrict__ Bt,
                                       __bf16* dA, __bf16* dB,
                                       int m0, int n0, int kt, int tid) {
  #pragma unroll
  for (int c = tid; c < 128 * 4; c += 256) {  // 4 async instr / thread
    const int row = c >> 2, off = (c & 3) << 3;
    async_copy16(&A[(size_t)(m0 + row) * DIM + kt + off], &dA[row * GLSA + off]);
    async_copy16(&Bt[(size_t)(n0 + row) * DIM + kt + off], &dB[row * GLSA + off]);
  }
}
#endif

template <int MODE>
__global__ void __launch_bounds__(256)
gemm_kernel(const __bf16* __restrict__ A, const __bf16* __restrict__ Bt,
            const float* __restrict__ bias, void* __restrict__ out) {
  constexpr int K = DIM, BM = 128;
  __shared__ alignas(16) __bf16 sA[2][BM * GLSA];
  __shared__ alignas(16) __bf16 sB[2][BM * GLSA];

  const int m0 = blockIdx.x * BM;
  const int n0 = blockIdx.y * BM;
  const int tid = threadIdx.x;
  const int w = tid >> 5, lane = tid & 31;
  const int wr0 = (w >> 1) * 32;   // wave row offset in tile
  const int wc0 = (w & 1) * 64;    // wave col offset in tile

  v8f acc[2][4];
  #pragma unroll
  for (int i = 0; i < 2; ++i)
    #pragma unroll
    for (int j = 0; j < 4; ++j) acc[i][j] = v8f{};

#if HAVE_ASYNC_LDS
  gemm_issue_tile(A, Bt, sA[0], sB[0], m0, n0, 0, tid);
  int cur = 0;
  for (int kt = 0; kt < K; kt += GBK) {
    const bool more = (kt + GBK < K);
    if (more) gemm_issue_tile(A, Bt, sA[cur ^ 1], sB[cur ^ 1], m0, n0, kt + GBK, tid);
    if (more) wait_async<4>();   // current tile done; next tile stays in flight
    else      wait_async<0>();
    __syncthreads();
    v16bf af0 = load_frag(&sA[cur][(wr0 + 0)  * GLSA], GLSA, 0);
    v16bf af1 = load_frag(&sA[cur][(wr0 + 16) * GLSA], GLSA, 0);
    #pragma unroll
    for (int ni = 0; ni < 4; ++ni) {
      v16bf bf = load_frag(&sB[cur][(wc0 + ni * 16) * GLSA], GLSA, 0);
      acc[0][ni] = wmma_bf16(af0, bf, acc[0][ni]);
      acc[1][ni] = wmma_bf16(af1, bf, acc[1][ni]);
    }
    __syncthreads();   // everyone done reading buf[cur] before it reloads
    cur ^= 1;
  }
#else
  for (int kt = 0; kt < K; kt += GBK) {
    __syncthreads();
    #pragma unroll
    for (int c = tid; c < BM * 4; c += 256) {
      const int row = c >> 2, off = (c & 3) << 3;
      *(uint4*)(&sA[0][row * GLSA + off]) =
          *(const uint4*)(&A[(size_t)(m0 + row) * K + kt + off]);
      *(uint4*)(&sB[0][row * GLSA + off]) =
          *(const uint4*)(&Bt[(size_t)(n0 + row) * K + kt + off]);
    }
    if (kt + GBK < K) {
      __builtin_prefetch(&A[(size_t)(m0 + (tid >> 1)) * K + kt + GBK], 0, 1);
      __builtin_prefetch(&Bt[(size_t)(n0 + (tid >> 1)) * K + kt + GBK], 0, 1);
    }
    __syncthreads();
    v16bf af0 = load_frag(&sA[0][(wr0 + 0)  * GLSA], GLSA, 0);
    v16bf af1 = load_frag(&sA[0][(wr0 + 16) * GLSA], GLSA, 0);
    #pragma unroll
    for (int ni = 0; ni < 4; ++ni) {
      v16bf bf = load_frag(&sB[0][(wc0 + ni * 16) * GLSA], GLSA, 0);
      acc[0][ni] = wmma_bf16(af0, bf, acc[0][ni]);
      acc[1][ni] = wmma_bf16(af1, bf, acc[1][ni]);
    }
  }
#endif

  // Epilogue. C layout: VGPR r, lane L -> row r + 8*(L/16), col L%16.
  #pragma unroll
  for (int mi = 0; mi < 2; ++mi)
    #pragma unroll
    for (int ni = 0; ni < 4; ++ni)
      #pragma unroll
      for (int r = 0; r < 8; ++r) {
        const int gm = m0 + wr0 + mi * 16 + r + ((lane >> 4) << 3);
        const int gn = n0 + wc0 + ni * 16 + (lane & 15);
        const float v = acc[mi][ni][r] + bias[gn];
        if constexpr (MODE == 2) {
          ((float*)out)[(size_t)gm * DIM + gn] = v;
        } else {
          const int b = gm >> 11, t = gm & (SEQ - 1);
          const int h = gn >> 7,  dh = gn & (DHD - 1);
          __bf16* o = (__bf16*)out;
          if constexpr (MODE == 0)
            o[(((size_t)b * NH + h) * SEQ + t) * DHD + dh] = (__bf16)v;
          else
            o[(((size_t)b * NH + h) * DHD + dh) * SEQ + t] = (__bf16)v;
        }
      }
}

// ---------------------------------------------------------------------------
// Flash attention (causal, online softmax).
// Grid: (T/64, B*H). Block: 128 threads = 4 waves; wave owns 16 q rows.
// Async path: K tile waited early (asynccnt<=8) so S=QK^T starts while the V
// tile is still in flight; V waited only right before the P*V WMMAs.
// ---------------------------------------------------------------------------
__global__ void __launch_bounds__(128)
flash_attn_kernel(const __bf16* __restrict__ qg, const __bf16* __restrict__ kg,
                  const __bf16* __restrict__ vtg, __bf16* __restrict__ og) {
  constexpr int BR = 64, BC = 64;
  constexpr int KS = DHD + 8;  // 136 elem (272B, 16B aligned)
  constexpr int VS = BC + 8;   // 72 elem
  constexpr int PS = BC + 8;   // 72 elem
  __shared__ alignas(16) __bf16 sK[BC * KS];        // K block  [s][d]
  __shared__ alignas(16) __bf16 sV[DHD * VS];       // V^T block [d][s]
  __shared__ alignas(16) __bf16 sP[4 * 16 * PS];    // per-wave P staging

  const int bh = blockIdx.y;  // b*NH + h
  const int qb = blockIdx.x;
  const int tid = threadIdx.x;
  const int w = tid >> 5, lane = tid & 31;
  const int q0 = qb * BR + w * 16;

  // Q strip 16x128 held in registers as 4 A-fragments (K windows of 32)
  const __bf16* qbase = qg + ((size_t)bh * SEQ + q0) * DHD;
  v16bf qf[4];
  #pragma unroll
  for (int kk = 0; kk < 4; ++kk) qf[kk] = load_frag(qbase, DHD, kk * 32);

  v8f accO[8];
  #pragma unroll
  for (int i = 0; i < 8; ++i) accO[i] = v8f{};
  float mrow[8], lrow[8];
  #pragma unroll
  for (int r = 0; r < 8; ++r) { mrow[r] = -__builtin_inff(); lrow[r] = 0.f; }

  const float scale = 0.08838834764831845f;  // 1/sqrt(128)

  for (int jb = 0; jb <= qb; ++jb) {
    const int j0 = jb * BC;
    __syncthreads();  // all waves done with previous sK/sV
#if HAVE_ASYNC_LDS
    #pragma unroll
    for (int c = tid; c < BC * (DHD / 8); c += 128) {  // K tile: 8 async/thr
      const int row = c >> 4, off = (c & 15) << 3;
      async_copy16(&kg[((size_t)bh * SEQ + j0 + row) * DHD + off],
                   &sK[row * KS + off]);
    }
    #pragma unroll
    for (int c = tid; c < DHD * (BC / 8); c += 128) {  // V tile: 8 async/thr
      const int row = c >> 3, off = (c & 7) << 3;
      async_copy16(&vtg[((size_t)bh * DHD + row) * SEQ + j0 + off],
                   &sV[row * VS + off]);
    }
    wait_async<8>();  // K tile complete; V tile still in flight
    __syncthreads();
#else
    #pragma unroll
    for (int c = tid; c < BC * (DHD / 8); c += 128) {
      const int row = c >> 4, off = (c & 15) << 3;
      *(uint4*)(&sK[row * KS + off]) =
          *(const uint4*)(&kg[((size_t)bh * SEQ + j0 + row) * DHD + off]);
    }
    #pragma unroll
    for (int c = tid; c < DHD * (BC / 8); c += 128) {
      const int row = c >> 3, off = (c & 7) << 3;
      *(uint4*)(&sV[row * VS + off]) =
          *(const uint4*)(&vtg[((size_t)bh * DHD + row) * SEQ + j0 + off]);
    }
    __syncthreads();
#endif

    // S = Q K^T  (16 x 64, f32) : 4 col tiles x 4 K-steps
    v8f s[4];
    #pragma unroll
    for (int ni = 0; ni < 4; ++ni) {
      s[ni] = v8f{};
      #pragma unroll
      for (int kk = 0; kk < 4; ++kk) {
        v16bf bf = load_frag(&sK[(ni * 16) * KS], KS, kk * 32);
        s[ni] = wmma_bf16(qf[kk], bf, s[ni]);
      }
    }

    // scale + causal mask on the diagonal block
    const bool diag = (jb == qb);
    #pragma unroll
    for (int ni = 0; ni < 4; ++ni)
      #pragma unroll
      for (int r = 0; r < 8; ++r) {
        float v = s[ni][r] * scale;
        if (diag) {
          const int col = j0 + ni * 16 + (lane & 15);
          const int row = q0 + r + ((lane >> 4) << 3);
          if (col > row) v = -__builtin_inff();
        }
        s[ni][r] = v;
      }

    // online softmax: row stats per-VGPR (row r+8*(lane/16)); reduce over the
    // 16 lanes holding the row's N values.
    float mnew[8], alpha[8];
    #pragma unroll
    for (int r = 0; r < 8; ++r) {
      float mx = fmaxf(fmaxf(s[0][r], s[1][r]), fmaxf(s[2][r], s[3][r]));
      #pragma unroll
      for (int d = 1; d < 16; d <<= 1) mx = fmaxf(mx, __shfl_xor(mx, d, 16));
      mnew[r] = fmaxf(mrow[r], mx);
      alpha[r] = __expf(mrow[r] - mnew[r]);
      mrow[r] = mnew[r];
    }
    #pragma unroll
    for (int ni = 0; ni < 4; ++ni)
      #pragma unroll
      for (int r = 0; r < 8; ++r) s[ni][r] = __expf(s[ni][r] - mnew[r]);
    #pragma unroll
    for (int r = 0; r < 8; ++r) {
      float sm = (s[0][r] + s[1][r]) + (s[2][r] + s[3][r]);
      #pragma unroll
      for (int d = 1; d < 16; d <<= 1) sm += __shfl_xor(sm, d, 16);
      lrow[r] = lrow[r] * alpha[r] + sm;
    }
    #pragma unroll
    for (int i = 0; i < 8; ++i)
      #pragma unroll
      for (int r = 0; r < 8; ++r) accO[i][r] *= alpha[r];

    // reshape P (C layout) -> A-fragment layout via wave-private LDS
    __bf16* pw = &sP[w * 16 * PS];
    #pragma unroll
    for (int ni = 0; ni < 4; ++ni)
      #pragma unroll
      for (int r = 0; r < 8; ++r) {
        const int row = r + ((lane >> 4) << 3);
        pw[row * PS + ni * 16 + (lane & 15)] = (__bf16)s[ni][r];
      }
    v16bf pf0 = load_frag(pw, PS, 0);   // same-wave LDS RAW: hw keeps in order
    v16bf pf1 = load_frag(pw, PS, 32);

#if HAVE_ASYNC_LDS
    wait_async<0>();   // our V-tile async loads complete
    __syncthreads();   // everyone's V tile visible
#endif

    // O += P V : 8 DH tiles x 2 K-steps
    #pragma unroll
    for (int dn = 0; dn < 8; ++dn) {
      v16bf bf0 = load_frag(&sV[(dn * 16) * VS], VS, 0);
      v16bf bf1 = load_frag(&sV[(dn * 16) * VS], VS, 32);
      accO[dn] = wmma_bf16(pf0, bf0, accO[dn]);
      accO[dn] = wmma_bf16(pf1, bf1, accO[dn]);
    }
  }

  // write merged-head output: [B,T,D] bf16
  const int b = bh >> 4, h = bh & (NH - 1);
  #pragma unroll
  for (int dn = 0; dn < 8; ++dn)
    #pragma unroll
    for (int r = 0; r < 8; ++r) {
      const int t = q0 + r + ((lane >> 4) << 3);
      const int d = dn * 16 + (lane & 15);
      og[((size_t)b * SEQ + t) * DIM + h * DHD + d] =
          (__bf16)(accO[dn][r] / lrow[r]);
    }
}

// ---------------------------------------------------------------------------
extern "C" void kernel_launch(void* const* d_in, const int* in_sizes, int n_in,
                              void* d_out, int out_size, void* d_ws,
                              size_t ws_size, hipStream_t stream) {
  const float* x  = (const float*)d_in[0];
  const float* Wq = (const float*)d_in[1];
  const float* bq = (const float*)d_in[2];
  const float* Wk = (const float*)d_in[3];
  const float* bk = (const float*)d_in[4];
  const float* Wv = (const float*)d_in[5];
  const float* bv = (const float*)d_in[6];
  const float* Wo = (const float*)d_in[7];
  const float* bo = (const float*)d_in[8];
  float* out = (float*)d_out;

  // Workspace partition (~192 MB; everything bf16 so q/k/v stay L2-resident)
  char* ws = (char*)d_ws;
  const size_t sz_x = (size_t)BB * SEQ * DIM * sizeof(__bf16);  // 32 MB
  const size_t sz_w = (size_t)DIM * DIM * sizeof(__bf16);       // 8 MB
  __bf16* xb  = (__bf16*)ws; ws += sz_x;
  __bf16* wqt = (__bf16*)ws; ws += sz_w;
  __bf16* wkt = (__bf16*)ws; ws += sz_w;
  __bf16* wvt = (__bf16*)ws; ws += sz_w;
  __bf16* wot = (__bf16*)ws; ws += sz_w;
  __bf16* q   = (__bf16*)ws; ws += sz_x;   // [B,H,T,DH]
  __bf16* k   = (__bf16*)ws; ws += sz_x;   // [B,H,T,DH]
  __bf16* vt  = (__bf16*)ws; ws += sz_x;   // [B,H,DH,T]
  __bf16* ao  = (__bf16*)ws; ws += sz_x;   // [B,T,D]

  // 1) fp32 -> bf16 conversions
  {
    const size_t n = (size_t)BB * SEQ * DIM;           // 16M elems
    cvt_kernel<<<dim3((unsigned)(n / (256 * 8))), dim3(256), 0, stream>>>(x, xb);
    dim3 tg(DIM / 32, DIM / 32), tb(32, 8);
    transpose_cvt_kernel<<<tg, tb, 0, stream>>>(Wq, wqt);
    transpose_cvt_kernel<<<tg, tb, 0, stream>>>(Wk, wkt);
    transpose_cvt_kernel<<<tg, tb, 0, stream>>>(Wv, wvt);
    transpose_cvt_kernel<<<tg, tb, 0, stream>>>(Wo, wot);
  }

  // 2) QKV projections (WMMA GEMMs)
  dim3 gg(BB * SEQ / 128, DIM / 128);  // (64, 16)
  gemm_kernel<0><<<gg, 256, 0, stream>>>(xb, wqt, bq, q);
  gemm_kernel<0><<<gg, 256, 0, stream>>>(xb, wkt, bk, k);
  gemm_kernel<1><<<gg, 256, 0, stream>>>(xb, wvt, bv, vt);

  // 3) causal flash attention
  flash_attn_kernel<<<dim3(SEQ / 64, BB * NH), 128, 0, stream>>>(q, k, vt, ao);

  // 4) output projection -> fp32 d_out
  gemm_kernel<2><<<gg, 256, 0, stream>>>(ao, wot, bo, out);
}
